// GBiNet_420906795162
// MI455X (gfx1250) — matrix operand order repaired and checked by reading, not compiled, not run on hardware
//
#include <hip/hip_runtime.h>
#include <hip/hip_bf16.h>

typedef __attribute__((ext_vector_type(16))) _Float16 v16h;
typedef __attribute__((ext_vector_type(8)))  float    v8f;

#define Vn 5
#define Cc 32
#define Hh 384
#define Ww 384
#define Dd 4
#define Gg 8
#define HWn (Hh*Ww)

// ---------------- setup: fold all camera algebra into A_s (3x3) + d_s (3) per view ----
__device__ __forceinline__ void inv3(const float* m, float* o) {
    float c00 =  (m[4]*m[8]-m[5]*m[7]);
    float c01 = -(m[3]*m[8]-m[5]*m[6]);
    float c02 =  (m[3]*m[7]-m[4]*m[6]);
    float det = m[0]*c00 + m[1]*c01 + m[2]*c02;
    float id = 1.0f/det;
    o[0]=c00*id; o[1]=-(m[1]*m[8]-m[2]*m[7])*id; o[2]= (m[1]*m[5]-m[2]*m[4])*id;
    o[3]=c01*id; o[4]= (m[0]*m[8]-m[2]*m[6])*id; o[5]=-(m[0]*m[5]-m[2]*m[3])*id;
    o[6]=c02*id; o[7]=-(m[0]*m[7]-m[1]*m[6])*id; o[8]= (m[0]*m[4]-m[1]*m[3])*id;
}
__device__ __forceinline__ void mm3(const float* a, const float* b, float* o) {
    for (int r=0;r<3;++r) for (int c=0;c<3;++c)
        o[r*3+c] = a[r*3+0]*b[0*3+c] + a[r*3+1]*b[1*3+c] + a[r*3+2]*b[2*3+c];
}
__device__ __forceinline__ void mv3(const float* a, const float* v, float* o) {
    for (int r=0;r<3;++r) o[r] = a[r*3+0]*v[0] + a[r*3+1]*v[1] + a[r*3+2]*v[2];
}

__global__ void gbinet_setup(const float* __restrict__ Ks, const float* __restrict__ Es,
                             float* __restrict__ proj) {
    if (threadIdx.x != 0 || blockIdx.x != 0) return;
    float K0inv[9], R0[9], t0[3], R0inv[9], M0[9], c0[3];
    inv3(Ks, K0inv);
    for (int r=0;r<3;++r) { for (int c=0;c<3;++c) R0[r*3+c] = Es[r*4+c]; t0[r] = Es[r*4+3]; }
    inv3(R0, R0inv);
    mm3(R0inv, K0inv, M0);
    mv3(R0inv, t0, c0);
    for (int s=1; s<Vn; ++s) {
        const float* Ksv = Ks + s*9;
        float Rs[9], ts[3];
        for (int r=0;r<3;++r) { for (int c=0;c<3;++c) Rs[r*3+c] = Es[s*12 + r*4+c]; ts[r] = Es[s*12 + r*4+3]; }
        float P[9], A[9], Pc0[3], Kt[3];
        mm3(Ksv, Rs, P);
        mm3(P, M0, A);
        mv3(P, c0, Pc0);
        mv3(Ksv, ts, Kt);
        float* o = proj + (s-1)*12;
        for (int k=0;k<9;++k) o[k] = A[k];
        for (int r=0;r<3;++r) o[9+r] = Kt[r] - Pc0[r];
    }
}

// ---------------- main kernel: gather + group correlation + WMMA MLP -------------------
__global__ __launch_bounds__(256) void gbinet_main(
    const float* __restrict__ feats, const float* __restrict__ depths,
    const float* __restrict__ w0, const float* __restrict__ b0,
    const float* __restrict__ w1, const float* __restrict__ b1,
    const float* __restrict__ w2, const float* __restrict__ b2,
    const float* __restrict__ proj, float* __restrict__ out)
{
    __shared__ float simLDS[8][16][Dd][Gg];   // per-wave sim tile [pixel][depth][group]
    __shared__ float stageLDS[8][16][16];     // per-wave MLP layer staging [m][n]
    __shared__ float vwLDS[8][16];            // per-wave pixel view-weights

    const int lane  = threadIdx.x & 31;
    const int wave  = threadIdx.x >> 5;
    const int p     = lane & 15;          // pixel within tile
    const int dHalf = lane >> 4;          // this lane's depth pair {dHalf, 2+dHalf}
    const int P     = (blockIdx.x*8 + wave)*16 + p;
    const int hh    = P / Ww;
    const int ww    = P - hh*Ww;
    const float fx  = (float)ww + 0.5f;
    const float fy  = (float)hh + 0.5f;
    const int  dIdx[2] = { dHalf, 2 + dHalf };
    float dep[2];
    dep[0] = depths[dIdx[0]*HWn + P];
    dep[1] = depths[dIdx[1]*HWn + P];
    const float* refB = feats + P;        // view 0, channel stride HWn

    // ---- MLP weights in WMMA B-matrix layout (f16, K zero-padded to 32) ----
    const int nCol = p;                   // N column this lane carries in B/D layouts
    v16h Bmat0, Bmat1;
    #pragma unroll
    for (int j=0;j<16;++j) { Bmat0[j]=(_Float16)0.0f; Bmat1[j]=(_Float16)0.0f; }
    if (lane < 16) {                      // lanes 0-15 carry K=0..15 (2 per VGPR); rest zero
        #pragma unroll
        for (int j=0;j<8;++j) Bmat0[j] = (_Float16)w0[nCol*8 + j];      // B0[k][n] = w0[n][k]
        if (nCol < 8) {
            #pragma unroll
            for (int j=0;j<16;++j) Bmat1[j] = (_Float16)w1[nCol*16 + j]; // B1[k][n] = w1[n][k]
        }
    }
    const float bias0 = b0[nCol];
    const float bias1 = (nCol < 8) ? b1[nCol] : 0.0f;
    float w2r[8];
    #pragma unroll
    for (int j=0;j<8;++j) w2r[j] = w2[j];
    const float b2r = b2[0];

    float accS[2][Gg];
    #pragma unroll
    for (int i=0;i<2;++i)
        #pragma unroll
        for (int g=0; g<Gg; ++g) accS[i][g] = 0.0f;
    float accW = 0.0f;

    for (int s=1; s<Vn; ++s) {
        const float* pr = proj + (s-1)*12;
        const float A00=pr[0],A01=pr[1],A02=pr[2],
                    A10=pr[3],A11=pr[4],A12=pr[5],
                    A20=pr[6],A21=pr[7],A22=pr[8];
        const float Dv0=pr[9],Dv1=pr[10],Dv2=pr[11];
        const float* src = feats + s*Cc*HWn;

        // ---- phase A: project + bilinear group correlation (2 depth tasks per lane) ----
        float simreg[2][Gg];
        #pragma unroll
        for (int i=0;i<2;++i) {
            const float dz = dep[i];
            const float u = (A00*fx + A01*fy + A02)*dz + Dv0;
            const float v = (A10*fx + A11*fy + A12)*dz + Dv1;
            const float z = (A20*fx + A21*fy + A22)*dz + Dv2 + 1e-9f;
            const float iz = 1.0f / z;
            float px = fminf(fmaxf(u*iz, -1.0e6f), 1.0e6f);
            float py = fminf(fmaxf(v*iz, -1.0e6f), 1.0e6f);
            const float x0f = floorf(px), y0f = floorf(py);
            const float wx1 = px - x0f, wx0 = 1.0f - wx1;
            const float wy1 = py - y0f, wy0 = 1.0f - wy1;
            const int x0 = (int)x0f, y0 = (int)y0f;
            float wt[4]; int off[4];
            #pragma unroll
            for (int t=0;t<4;++t) {
                const int tx = t & 1, ty = t >> 1;
                const float xf = x0f + (float)tx;
                const float yf = y0f + (float)ty;
                const bool valid = (xf >= 0.0f) && (xf <= (float)(Ww-1)) &&
                                   (yf >= 0.0f) && (yf <= (float)(Hh-1));
                const int xc = min(max(x0+tx, 0), Ww-1);
                const int yc = min(max(y0+ty, 0), Hh-1);
                off[t] = yc*Ww + xc;
                const float wxy = (tx ? wx1 : wx0) * (ty ? wy1 : wy0);
                wt[t] = valid ? wxy : 0.0f;
            }
            #pragma unroll
            for (int g=0; g<Gg; ++g) {
                float acc = 0.0f;
                #pragma unroll
                for (int cc=0; cc<4; ++cc) {
                    const int c = g*4 + cc;
                    const float* fc = src + c*HWn;
                    const float wv = wt[0]*fc[off[0]] + wt[1]*fc[off[1]]
                                   + wt[2]*fc[off[2]] + wt[3]*fc[off[3]];
                    acc += wv * refB[c*HWn];
                }
                simreg[i][g] = acc * 0.25f;                       // mean over C/G=4
                simLDS[wave][p][dIdx[i]][g] = simreg[i][g];
            }
        }
        asm volatile("s_wait_dscnt 0" ::: "memory");   // wave-local: sim tile visible

        // ---- phase B: per-depth MLP via WMMA (16 pixels = M rows) ----
        float maxlogit = -3.0e38f;
        for (int dd=0; dd<Dd; ++dd) {
            // A0 (16x32 f16, A-layout): lanes0-15 carry M=lane, K=0..7 in elems 0..7;
            // lanes16-31 carry K=8..15 (zero). K=16..31 zero.
            v16h A0;
            #pragma unroll
            for (int j=0;j<16;++j) A0[j] = (_Float16)0.0f;
            if (lane < 16) {
                #pragma unroll
                for (int j=0;j<8;++j) A0[j] = (_Float16)simLDS[wave][lane][dd][j];
            }
            v8f Cz;
            #pragma unroll
            for (int j=0;j<8;++j) Cz[j] = 0.0f;
            v8f O0 = __builtin_amdgcn_wmma_f32_16x16x32_f16(false, A0, false, Bmat0,
                                                            (short)0, Cz, false, false);
            // D-layout -> LDS [m][n] with bias+relu
            const int mOff = (lane < 16) ? 0 : 8;
            #pragma unroll
            for (int r=0;r<8;++r) stageLDS[wave][mOff + r][nCol] = fmaxf(O0[r] + bias0, 0.0f);
            asm volatile("s_wait_dscnt 0" ::: "memory");

            // A1 (hidden 16 wide, zero-padded to K=32): lane<16 reads K=0..15, lane>=16 zero
            v16h A1;
            #pragma unroll
            for (int j=0;j<16;++j) A1[j] = (_Float16)0.0f;
            const int kb = (lane < 16) ? 0 : 8;
            #pragma unroll
            for (int j=0;j<16;++j) if (kb == 0) A1[j] = (_Float16)stageLDS[wave][p][j];
            v8f O1 = __builtin_amdgcn_wmma_f32_16x16x32_f16(false, A1, false, Bmat1,
                                                            (short)0, Cz, false, false);
            #pragma unroll
            for (int r=0;r<8;++r) stageLDS[wave][mOff + r][nCol] = fmaxf(O1[r] + bias1, 0.0f);
            asm volatile("s_wait_dscnt 0" ::: "memory");

            // layer 2 (8->1) scalar on lanes 0-15; track max logit over depth
            if (lane < 16) {
                float logit = b2r;
                #pragma unroll
                for (int j=0;j<8;++j) logit += w2r[j]*stageLDS[wave][lane][j];
                maxlogit = fmaxf(maxlogit, logit);
            }
            asm volatile("s_wait_dscnt 0" ::: "memory");
        }
        if (lane < 16) vwLDS[wave][lane] = 1.0f/(1.0f + expf(-maxlogit)); // sigmoid(max)=max(sigmoid)
        asm volatile("s_wait_dscnt 0" ::: "memory");
        const float vw = vwLDS[wave][p];
        #pragma unroll
        for (int i=0;i<2;++i)
            #pragma unroll
            for (int g=0; g<Gg; ++g) accS[i][g] += simreg[i][g]*vw;
        accW += vw;
    }

    const float invw = 1.0f/accW;
    #pragma unroll
    for (int i=0;i<2;++i) {
        const int dd = dIdx[i];
        #pragma unroll
        for (int g=0; g<Gg; ++g)
            out[(g*Dd + dd)*HWn + P] = accS[i][g]*invw;
    }
}

extern "C" void kernel_launch(void* const* d_in, const int* in_sizes, int n_in,
                              void* d_out, int out_size, void* d_ws, size_t ws_size,
                              hipStream_t stream) {
    (void)in_sizes; (void)n_in; (void)out_size; (void)ws_size;
    const float* feats  = (const float*)d_in[0];
    const float* depths = (const float*)d_in[1];
    const float* Ks     = (const float*)d_in[2];
    const float* Es     = (const float*)d_in[3];
    const float* w0     = (const float*)d_in[4];
    const float* b0     = (const float*)d_in[5];
    const float* w1     = (const float*)d_in[6];
    const float* b1     = (const float*)d_in[7];
    const float* w2     = (const float*)d_in[8];
    const float* b2     = (const float*)d_in[9];
    float* proj = (float*)d_ws;           // 4 views x 12 floats
    float* outp = (float*)d_out;

    gbinet_setup<<<1, 32, 0, stream>>>(Ks, Es, proj);
    // 147456 pixels / (16 px per wave * 8 waves per block) = 1152 blocks
    gbinet_main<<<1152, 256, 0, stream>>>(feats, depths, w0, b0, w1, b1, w2, b2, proj, outp);
}
